// SparseTriangleCrossMultiplicativeUpdate_56023553409071
// MI455X (gfx1250) — matrix operand order repaired and checked by reading, not compiled, not run on hardware
//
#include <hip/hip_runtime.h>
#include <hip/hip_bf16.h>
#include <math.h>

#define N_NODES 2048
#define E_EDGES 32768
#define EE_PAIRS 262144
#define CS 384
#define CZ 128
#define CG 16
#define NRBF 64

typedef __attribute__((ext_vector_type(16))) _Float16 v16h;
typedef __attribute__((ext_vector_type(8)))  float    v8f;

// ---------------------------------------------------------------------------
// Kernel 0: zero the [E, CZ] float accumulator (must be re-zeroed each launch)
// ---------------------------------------------------------------------------
__global__ void zero_acc_kernel(float* __restrict__ acc, int n) {
    int i = blockIdx.x * blockDim.x + threadIdx.x;
    if (i < n) acc[i] = 0.0f;
}

// ---------------------------------------------------------------------------
// Kernel 1: pack w_dg [256,128] and w_dp [64,128] (row-major [in,out]) into
// WMMA-B wave32 fragment layout: frag index ((kt*8+nt)*32+lane)*16 + t
//   K = kt*32 + (lane>=16 ? 8 : 0) + (t&7) + (t>>3)*16 ,  N = nt*16 + (lane&15)
// ---------------------------------------------------------------------------
__global__ void pack_b_kernel(const float* __restrict__ w_dg,
                              const float* __restrict__ w_dp,
                              _Float16* __restrict__ Bdg,
                              _Float16* __restrict__ Bdp) {
    int id = blockIdx.x * blockDim.x + threadIdx.x;
    if (id < 8 * 8 * 32 * 16) {                 // w_dg: 8 K-tiles
        int t    = id & 15;
        int lane = (id >> 4) & 31;
        int nt   = (id >> 9) & 7;
        int kt   = id >> 12;
        int K = kt * 32 + ((lane >= 16) ? 8 : 0) + (t & 7) + (t >> 3) * 16;
        int Nc = nt * 16 + (lane & 15);
        Bdg[id] = (_Float16)w_dg[K * CZ + Nc];
    } else if (id < 8 * 8 * 32 * 16 + 2 * 8 * 32 * 16) {  // w_dp: 2 K-tiles
        int id2  = id - 8 * 8 * 32 * 16;
        int t    = id2 & 15;
        int lane = (id2 >> 4) & 31;
        int nt   = (id2 >> 9) & 7;
        int kt   = id2 >> 12;
        int K = kt * 32 + ((lane >= 16) ? 8 : 0) + (t & 7) + (t >> 3) * 16;
        int Nc = nt * 16 + (lane & 15);
        Bdp[id2] = (_Float16)w_dp[K * CZ + Nc];
    }
}

// ---------------------------------------------------------------------------
// Kernel 2: node projections  left = nf@w_nl+b_nl, right = nf@w_nr+b_nr
// stored as f16 rows of 16 (feed the WMMA-A outer product directly)
// ---------------------------------------------------------------------------
__global__ void node_proj_kernel(const float* __restrict__ nf,
                                 const float* __restrict__ w_nl,
                                 const float* __restrict__ b_nl,
                                 const float* __restrict__ w_nr,
                                 const float* __restrict__ b_nr,
                                 _Float16* __restrict__ left_h,
                                 _Float16* __restrict__ right_h) {
    int id = blockIdx.x * blockDim.x + threadIdx.x;
    if (id >= N_NODES * CG) return;
    int row = id >> 4;
    int col = id & 15;
    const float* x = nf + (size_t)row * CS;
    float sl = b_nl[col];
    float sr = b_nr[col];
    #pragma unroll 4
    for (int k = 0; k < CS; ++k) {
        float xv = x[k];
        sl += xv * w_nl[k * CG + col];
        sr += xv * w_nr[k * CG + col];
    }
    left_h[id]  = (_Float16)sl;
    right_h[id] = (_Float16)sr;
}

// ---------------------------------------------------------------------------
// shared-memory LayerNorm helper over a 128-wide row (blockDim.x == 128)
// ---------------------------------------------------------------------------
__device__ __forceinline__ float row_layernorm(float x, float g, float b,
                                               float* red, int t) {
    red[t] = x;
    __syncthreads();
    for (int o = 64; o > 0; o >>= 1) {
        if (t < o) red[t] += red[t + o];
        __syncthreads();
    }
    float mean = red[0] * (1.0f / 128.0f);
    __syncthreads();
    float d = x - mean;
    red[t] = d * d;
    __syncthreads();
    for (int o = 64; o > 0; o >>= 1) {
        if (t < o) red[t] += red[t + o];
        __syncthreads();
    }
    float var = red[0] * (1.0f / 128.0f);
    __syncthreads();
    return d * rsqrtf(var + 1e-5f) * g + b;
}

// ---------------------------------------------------------------------------
// Kernel 3: edge2[e] = sigmoid(LN(src_ef)@w_eg+b_eg) * (LN(src_ef)@w_ep+b_ep)
// one 128-thread block per edge row
// ---------------------------------------------------------------------------
__global__ void edge2_kernel(const float* __restrict__ src_ef,
                             const float* __restrict__ g_src,
                             const float* __restrict__ bt_src,
                             const float* __restrict__ w_eg,
                             const float* __restrict__ b_eg,
                             const float* __restrict__ w_ep,
                             const float* __restrict__ b_ep,
                             float* __restrict__ edge2) {
    __shared__ float row[128];
    __shared__ float red[128];
    int t = threadIdx.x;
    int e = blockIdx.x;
    float x = src_ef[(size_t)e * CZ + t];
    float ln = row_layernorm(x, g_src[t], bt_src[t], red, t);
    row[t] = ln;
    __syncthreads();
    float sg = b_eg[t];
    float sp = b_ep[t];
    #pragma unroll 4
    for (int k = 0; k < CZ; ++k) {
        float v = row[k];
        sg += v * w_eg[k * CZ + t];
        sp += v * w_ep[k * CZ + t];
    }
    edge2[(size_t)e * CZ + t] = (1.0f / (1.0f + __expf(-sg))) * sp;
}

// ---------------------------------------------------------------------------
// Kernel 4 (dominant): fused per-pair outer-product GEMM + RBF GEMM via WMMA,
// gate/sigmoid combine, edge2 gather, atomic segment-sum into acc[E,CZ].
// One wave32 handles a tile of 16 edge-edge pairs x 128 output columns.
// ---------------------------------------------------------------------------
__global__ void __launch_bounds__(256)
fused_ee_kernel(const int* __restrict__ eeidx,     // [2,EE]
                const int* __restrict__ src_eidx,  // [2,E]
                const int* __restrict__ dst_eidx,  // [2,E]
                const float* __restrict__ trans,   // [N,3]
                const _Float16* __restrict__ left_h,
                const _Float16* __restrict__ right_h,
                const _Float16* __restrict__ Bdg,
                const _Float16* __restrict__ Bdp,
                const float* __restrict__ b_dg,
                const float* __restrict__ b_dp,
                const float* __restrict__ edge2,
                float* __restrict__ acc) {
    const int lane = threadIdx.x & 31;
    const int wv   = threadIdx.x >> 5;
    const int tile = blockIdx.x * (blockDim.x >> 5) + wv;
    const int m    = lane & 15;
    const int hi   = lane >> 4;          // K-half selector in A/B layouts
    const int cl   = lane & 15;          // output column within N-tile

    const int pair = tile * 16 + m;
    const int ee0m = eeidx[pair];                 // dst-edge index of pair m
    const int ee1m = eeidx[EE_PAIRS + pair];      // src-edge index of pair m
    const int n1   = src_eidx[ee1m];              // src_edge_index[0][ee1]
    const int n2   = dst_eidx[ee0m];              // dst_edge_index[0][ee0]

    const v16h lh = *(const v16h*)(left_h  + n1 * CG);
    const v16h rh = *(const v16h*)(right_h + n2 * CG);

    // distance for RBF (per pair m, replicated in both lane halves)
    float dx = trans[n1 * 3 + 0] - trans[n2 * 3 + 0] + 1e-8f;
    float dy = trans[n1 * 3 + 1] - trans[n2 * 3 + 1] + 1e-8f;
    float dz = trans[n1 * 3 + 2] - trans[n2 * 3 + 2] + 1e-8f;
    float dist = sqrtf(dx * dx + dy * dy + dz * dz);

    // 32-bit element offsets of the 8 output rows this lane touches (M=r+8*hi)
    int off_r[8];
    #pragma unroll
    for (int r = 0; r < 8; ++r)
        off_r[r] = eeidx[tile * 16 + hi * 8 + r] * CZ;

    // biases for this lane's output column in each of the 8 N-tiles
    float bg8[8], bp8[8];
    #pragma unroll
    for (int nt = 0; nt < 8; ++nt) {
        bg8[nt] = b_dg[nt * 16 + cl];
        bp8[nt] = b_dp[nt * 16 + cl];
    }

    // Build WMMA-A fragments for the outer-product matrix (K = 256, 8 tiles).
    // ISA 16-bit A 16x32 layout: element t -> K_local = (t>>3)*16 + (t&7) + 8*hi
    // so global k = kt*32 + K_local, i = k>>4 = 2*kt + (t>>3), j = (t&7) + 8*hi
    v16h a_op[8];
    #pragma unroll
    for (int kt = 0; kt < 8; ++kt) {
        #pragma unroll
        for (int t = 0; t < 16; ++t) {
            a_op[kt][t] = lh[2 * kt + (t >> 3)] * rh[(t & 7) + 8 * hi];
        }
    }

    // RBF A fragments (K = 64, 2 tiles): mu_k = k*20/63, sigma = 20/64
    v16h a_rbf[2];
    #pragma unroll
    for (int kt = 0; kt < 2; ++kt) {
        #pragma unroll
        for (int t = 0; t < 16; ++t) {
            int k = kt * 32 + (t >> 3) * 16 + (t & 7) + 8 * hi;
            float mu = (float)k * (20.0f / 63.0f);
            float rr = (dist - mu) * (64.0f / 20.0f);
            a_rbf[kt][t] = (_Float16)__expf(-rr * rr);
        }
    }

    #pragma unroll 1
    for (int nt = 0; nt < 8; ++nt) {
        // per-N-tile fragment bases; kt stride = 4096 halves (8 KB)
        const _Float16* pg = Bdg + nt * 512 + lane * 16;
        const _Float16* pp = Bdp + nt * 512 + lane * 16;

        // gate = (outer ⊗) @ w_dg : 8 K-steps of 32, B double-buffered
        v8f cg = {};
        v16h bcur = *(const v16h*)(pg);
        #pragma unroll
        for (int kt = 0; kt < 8; ++kt) {
            v16h bnext;
            if (kt < 7) bnext = *(const v16h*)(pg + (kt + 1) * 4096);
            else        bnext = *(const v16h*)(pp);          // first dist frag
            cg = __builtin_amdgcn_wmma_f32_16x16x32_f16(
                false, a_op[kt], false, bcur, (short)0, cg, false, false);
            bcur = bnext;
        }
        // dist_feat = rbf @ w_dp : 2 K-steps of 32
        v16h b2 = *(const v16h*)(pp + 4096);
        v8f cd = {};
        cd = __builtin_amdgcn_wmma_f32_16x16x32_f16(
            false, a_rbf[0], false, bcur, (short)0, cd, false, false);
        cd = __builtin_amdgcn_wmma_f32_16x16x32_f16(
            false, a_rbf[1], false, b2, (short)0, cd, false, false);

        const int col = nt * 16 + cl;
        const float bg = bg8[nt];
        const float bp = bp8[nt];
        // C layout: VGPR r, lane l -> row M = r + 8*hi, col = lane&15 (+16*nt)
        #pragma unroll
        for (int r = 0; r < 8; ++r) {
            int o = off_r[r] + col;                 // 32-bit element offset
            float gate = 1.0f / (1.0f + __expf(-(cg[r] + bg)));
            float val  = gate * (cd[r] + bp) * edge2[o];
            atomicAdd(acc + o, val);
        }
    }
}

// ---------------------------------------------------------------------------
// Kernel 5: out[e] = (LN(acc[e])@w_lo + b_lo) * sigmoid(LN(dst_ef[e])@w_og+b_og)
// one 128-thread block per edge row
// ---------------------------------------------------------------------------
__global__ void finalize_kernel(const float* __restrict__ acc,
                                const float* __restrict__ dst_ef,
                                const float* __restrict__ g_out,
                                const float* __restrict__ bt_out,
                                const float* __restrict__ w_lo,
                                const float* __restrict__ b_lo,
                                const float* __restrict__ g_dst,
                                const float* __restrict__ bt_dst,
                                const float* __restrict__ w_og,
                                const float* __restrict__ b_og,
                                float* __restrict__ out) {
    __shared__ float rowa[128];
    __shared__ float rowz[128];
    __shared__ float red[128];
    int t = threadIdx.x;
    int e = blockIdx.x;
    float a = acc[(size_t)e * CZ + t];
    float z = dst_ef[(size_t)e * CZ + t];
    float lna = row_layernorm(a, g_out[t], bt_out[t], red, t);
    __syncthreads();
    float lnz = row_layernorm(z, g_dst[t], bt_dst[t], red, t);
    rowa[t] = lna;
    rowz[t] = lnz;
    __syncthreads();
    float so = b_lo[t];
    float sg = b_og[t];
    #pragma unroll 4
    for (int k = 0; k < CZ; ++k) {
        so += rowa[k] * w_lo[k * CZ + t];
        sg += rowz[k] * w_og[k * CZ + t];
    }
    out[(size_t)e * CZ + t] = so * (1.0f / (1.0f + __expf(-sg)));
}

// ---------------------------------------------------------------------------
extern "C" void kernel_launch(void* const* d_in, const int* in_sizes, int n_in,
                              void* d_out, int out_size, void* d_ws, size_t ws_size,
                              hipStream_t stream) {
    (void)in_sizes; (void)n_in; (void)out_size; (void)ws_size;

    const float* node_features = (const float*)d_in[0];
    const float* node_trans    = (const float*)d_in[1];
    const float* dst_ef        = (const float*)d_in[2];
    const int*   dst_eidx      = (const int*)d_in[3];
    const float* src_ef        = (const float*)d_in[4];
    const int*   src_eidx      = (const int*)d_in[5];
    const int*   ee_idx        = (const int*)d_in[6];
    const float* w_nl  = (const float*)d_in[7];
    const float* b_nl  = (const float*)d_in[8];
    const float* w_nr  = (const float*)d_in[9];
    const float* b_nr  = (const float*)d_in[10];
    const float* g_dst = (const float*)d_in[11];
    const float* bt_dst= (const float*)d_in[12];
    const float* g_src = (const float*)d_in[13];
    const float* bt_src= (const float*)d_in[14];
    const float* w_ep  = (const float*)d_in[15];
    const float* b_ep  = (const float*)d_in[16];
    const float* w_eg  = (const float*)d_in[17];
    const float* b_eg  = (const float*)d_in[18];
    const float* w_dg  = (const float*)d_in[19];
    const float* b_dg  = (const float*)d_in[20];
    const float* w_dp  = (const float*)d_in[21];
    const float* b_dp  = (const float*)d_in[22];
    const float* g_out = (const float*)d_in[23];
    const float* bt_out= (const float*)d_in[24];
    const float* w_lo  = (const float*)d_in[25];
    const float* b_lo  = (const float*)d_in[26];
    const float* w_og  = (const float*)d_in[27];
    const float* b_og  = (const float*)d_in[28];
    float* out = (float*)d_out;

    // workspace layout (bytes)
    char* ws = (char*)d_ws;
    _Float16* left_h  = (_Float16*)(ws + 0);                        // 64 KB
    _Float16* right_h = (_Float16*)(ws + (64 << 10));               // 64 KB
    _Float16* Bdg     = (_Float16*)(ws + (128 << 10));              // 64 KB
    _Float16* Bdp     = (_Float16*)(ws + (192 << 10));              // 16 KB
    float*    edge2   = (float*)(ws + (256 << 10));                 // 16 MB
    float*    acc     = (float*)(ws + (256 << 10) + (size_t)E_EDGES * CZ * 4); // 16 MB

    // 1) pack WMMA-B operands (w_dg, w_dp)
    {
        int total = 8 * 8 * 32 * 16 + 2 * 8 * 32 * 16;   // 40960
        pack_b_kernel<<<(total + 255) / 256, 256, 0, stream>>>(w_dg, w_dp, Bdg, Bdp);
    }
    // 2) node projections -> f16
    {
        int total = N_NODES * CG;                         // 32768
        node_proj_kernel<<<(total + 255) / 256, 256, 0, stream>>>(
            node_features, w_nl, b_nl, w_nr, b_nr, left_h, right_h);
    }
    // 3) edge2 from src edge features
    edge2_kernel<<<E_EDGES, 128, 0, stream>>>(
        src_ef, g_src, bt_src, w_eg, b_eg, w_ep, b_ep, edge2);
    // 4) zero accumulator
    {
        int total = E_EDGES * CZ;                         // 4 M
        zero_acc_kernel<<<(total + 255) / 256, 256, 0, stream>>>(acc, total);
    }
    // 5) dominant fused WMMA kernel: 16 pairs per wave, 8 waves per block
    {
        int tiles = EE_PAIRS / 16;                        // 16384
        fused_ee_kernel<<<tiles / 8, 256, 0, stream>>>(
            ee_idx, src_eidx, dst_eidx, node_trans, left_h, right_h,
            Bdg, Bdp, b_dg, b_dp, edge2, acc);
    }
    // 6) finalize: LN + output projection + sigmoid gate
    finalize_kernel<<<E_EDGES, 128, 0, stream>>>(
        acc, dst_ef, g_out, bt_out, w_lo, b_lo, g_dst, bt_dst, w_og, b_og, out);
}
